// CorrLSTM_upsample_avg_29609504538953
// MI455X (gfx1250) — compile-verified, matrix operand order checked by reading, not run on hardware
//
#include <hip/hip_runtime.h>
#include <hip/hip_bf16.h>

typedef __attribute__((ext_vector_type(16))) __bf16 v16bf;
typedef __attribute__((ext_vector_type(8)))  float  v8f;
typedef __attribute__((ext_vector_type(4)))  int    v4i;

#if defined(__has_builtin)
#if __has_builtin(__builtin_amdgcn_global_load_async_to_lds_b128)
#define ASYNC_LDS 1
#endif
#endif

__device__ __forceinline__ float leaky01(float x) { return x > 0.f ? x : 0.01f * x; }

// ---------------------------------------------------------------------------
// 1) Composite per-axis pooling weights for k stacked (3, stride2, pad1) pools.
// ---------------------------------------------------------------------------
__global__ void comp_weights_k(float* __restrict__ w, int k) {
  if (blockIdx.x != 0 || threadIdx.x != 0) return;
  float cur[128];
  int half = 1;
  cur[0] = cur[1] = cur[2] = 1.f / 3.f;
  for (int j = 1; j < k; ++j) {
    int nh = 2 * half + 1;
    float nxt[128];
    for (int q = -nh; q <= nh; ++q) {
      float s = 0.f;
      for (int t = -1; t <= 1; ++t) {
        int o = q - (half + 1) * t;
        if (o >= -half && o <= half) s += cur[o + half];
      }
      nxt[q + nh] = s * (1.f / 3.f);
    }
    half = nh;
    for (int i = 0; i < 2 * half + 1; ++i) cur[i] = nxt[i];
  }
  for (int i = 0; i < 2 * half + 1; ++i) w[i] = cur[i];
}

// ---------------------------------------------------------------------------
// 2) Weight table A[d][Yo][r in {-1,0}][sy in {0,1,2}].
// ---------------------------------------------------------------------------
__global__ void atab_k(float* __restrict__ atab, const float* __restrict__ w,
                       int SZ, int lf) {
  int tid = blockIdx.x * blockDim.x + threadIdx.x;
  if (tid >= 5 * SZ) return;
  int d = tid / SZ, Yo = tid - d * SZ;
  int di = d - 2;
  int f = 1 << lf;
  int fine = SZ << lf;
  float A[2][3] = {{0.f,0.f,0.f},{0.f,0.f,0.f}};
  int base = f * Yo - (f - 1);
  for (int o = 0; o < 2 * f - 1; ++o) {
    int y = base + o;
    if (y < 0 || y >= fine) continue;
    float wv = w[o];
    int Y = y >> lf;
    int r = Y - Yo;
    int q = y + di;
    int Yq = (q < 0) ? -1 : (q >> lf);
    int sy = Yq - Y + 1;
    A[r + 1][sy] += wv;
  }
  float* out = atab + tid * 6;
  #pragma unroll
  for (int r = 0; r < 2; ++r)
    #pragma unroll
    for (int s = 0; s < 3; ++s) out[r * 3 + s] = A[r][s];
}

// ---------------------------------------------------------------------------
// 3) Coarse 9-point correlation: L[b][sy*3+sx][Y][X] = leaky(sum_c f1*f2_nbr)
// ---------------------------------------------------------------------------
__global__ void dot_k(float* __restrict__ Ld, const float* __restrict__ f1,
                      const float* __restrict__ f2, int CH, int SZ, int szsh) {
  int tid = blockIdx.x * blockDim.x + threadIdx.x;
  int SZ2 = 1 << (2 * szsh);
  if (tid >= 2 * SZ2) return;
  int b  = tid >> (2 * szsh);
  int sp = tid & (SZ2 - 1);
  int Y = sp >> szsh, X = sp & (SZ - 1);
  const float* p1 = f1 + (size_t)b * CH * SZ2 + sp;
  const float* p2 = f2 + (size_t)b * CH * SZ2 + sp;
  int   doff[9];
  float msk[9];
  #pragma unroll
  for (int isy = 0; isy < 3; ++isy)
    #pragma unroll
    for (int isx = 0; isx < 3; ++isx) {
      int s = isy * 3 + isx;
      int Yn = Y + isy - 1, Xn = X + isx - 1;
      bool v = (Yn >= 0) && (Yn < SZ) && (Xn >= 0) && (Xn < SZ);
      doff[s] = v ? ((isy - 1) << szsh) + (isx - 1) : 0;
      msk[s]  = v ? 1.f : 0.f;
    }
  float acc[9] = {0.f,0.f,0.f,0.f,0.f,0.f,0.f,0.f,0.f};
  for (int c = 0; c < CH; ++c) {
    float v1 = p1[(size_t)c * SZ2];
    const float* pc = p2 + (size_t)c * SZ2;
    #pragma unroll
    for (int s = 0; s < 9; ++s) acc[s] += v1 * pc[doff[s]];
  }
  #pragma unroll
  for (int s = 0; s < 9; ++s)
    Ld[((size_t)(b * 9 + s) << (2 * szsh)) + sp] = leaky01(acc[s] * msk[s]);
}

// ---------------------------------------------------------------------------
// 4) Fused expand+leaky+pool-chain (exact) + concat with h -> bf16 Feat.
// ---------------------------------------------------------------------------
__global__ void pooled_k(unsigned short* __restrict__ feat,
                         const float* __restrict__ Ld,
                         const float* __restrict__ atab,
                         const float* __restrict__ hin,
                         int SZ, int szsh, int total) {
  int tid = blockIdx.x * blockDim.x + threadIdx.x;
  if (tid >= total) return;
  int xix = tid & (SZ - 1);
  int Yo  = (tid >> szsh) & (SZ - 1);
  int bc  = tid >> (2 * szsh);
  int ch  = bc % 50;
  int b   = bc / 50;
  float val;
  if (ch >= 25) {
    val = hin[((size_t)(b * 25 + (ch - 25)) << (2 * szsh)) + (Yo << szsh) + xix];
  } else {
    int pi = ch / 5, pj = ch - pi * 5;
    float acc = 0.f;
    #pragma unroll
    for (int ry = -1; ry <= 0; ++ry) {
      int Yc = Yo + ry;
      if (Yc < 0) continue;
      const float* ayp = atab + ((pi * SZ + Yo) * 2 + (ry + 1)) * 3;
      #pragma unroll
      for (int rx = -1; rx <= 0; ++rx) {
        int Xc = xix + rx;
        if (Xc < 0) continue;
        const float* axp = atab + ((pj * SZ + xix) * 2 + (rx + 1)) * 3;
        const float* Lp = Ld + ((size_t)(b * 9) << (2 * szsh)) + (Yc << szsh) + Xc;
        #pragma unroll
        for (int sy = 0; sy < 3; ++sy) {
          float ay = ayp[sy];
          #pragma unroll
          for (int sx = 0; sx < 3; ++sx)
            acc += ay * axp[sx] * Lp[(size_t)(sy * 3 + sx) << (2 * szsh)];
        }
      }
    }
    val = acc;
  }
  __hip_bfloat16 bv = __float2bfloat16(val);
  feat[tid] = *reinterpret_cast<unsigned short*>(&bv);
}

// ---------------------------------------------------------------------------
// 5) Conv weights fp32 [100][50][3][3] -> bf16 GEMM matrix [112][480].
// ---------------------------------------------------------------------------
__global__ void wprep_k(unsigned short* __restrict__ wb, const float* __restrict__ W) {
  int tid = blockIdx.x * blockDim.x + threadIdx.x;
  if (tid >= 112 * 480) return;
  int m = tid / 480, k = tid - m * 480;
  float v = 0.f;
  if (m < 100 && k < 450) {
    int ic = k / 9, t = k - ic * 9, ky = t / 3, kx = t - ky * 3;
    v = W[((m * 50 + ic) * 3 + ky) * 3 + kx];
  }
  __hip_bfloat16 bv = __float2bfloat16(v);
  wb[tid] = *reinterpret_cast<unsigned short*>(&bv);
}

// ---------------------------------------------------------------------------
// 6) im2col pack: Col[kq][n][8] bf16, kq = k/8 (60 groups of 8), n < Npad.
//    Columns n >= N and K >= 450 are zero-filled so the GEMM needs no guards.
// ---------------------------------------------------------------------------
__global__ void pack_k(unsigned short* __restrict__ Col,
                       const unsigned short* __restrict__ feat,
                       int SZ, int szsh, int N, int Npad) {
  int tid = blockIdx.x * blockDim.x + threadIdx.x;
  if (tid >= 60 * Npad) return;
  int n  = tid % Npad;
  int kq = tid / Npad;
  unsigned short u[8];
  if (n >= N) {
    #pragma unroll
    for (int j = 0; j < 8; ++j) u[j] = 0;
  } else {
    int SZ2 = 1 << (2 * szsh);
    int b  = n >> (2 * szsh);
    int sp = n & (SZ2 - 1);
    int y = sp >> szsh, x = sp & (SZ - 1);
    const unsigned short* fb = feat + ((size_t)(b * 50) << (2 * szsh));
    #pragma unroll
    for (int j = 0; j < 8; ++j) {
      int k = kq * 8 + j;
      unsigned short v = 0;
      if (k < 450) {
        int ic = k / 9;
        int t = k - ic * 9;
        int ky = t / 3, kx = t - ky * 3;
        int yy = y + ky - 1, xx = x + kx - 1;
        if ((unsigned)yy < (unsigned)SZ && (unsigned)xx < (unsigned)SZ)
          v = fb[((size_t)ic << (2 * szsh)) + (yy << szsh) + xx];
      }
      u[j] = v;
    }
  }
  uint4* dst = reinterpret_cast<uint4*>(Col + ((size_t)kq * Npad + n) * 8);
  *dst = *reinterpret_cast<const uint4*>(u);
}

// ---------------------------------------------------------------------------
// 7) GEMM D[112, Npad] = Wb[112,480] x Col via v_wmma_f32_16x16x32_bf16.
//    A panel staged in LDS (3 chunks of 112x160 bf16, shared by 8 waves);
//    B loads are coalesced global b128 from the packed Col.
// ---------------------------------------------------------------------------
__global__ __launch_bounds__(256) void gemm_k(float* __restrict__ gates,
                                              const unsigned short* __restrict__ Wb,
                                              const unsigned short* __restrict__ Col,
                                              int Npad, int N) {
  __shared__ uint4 lAq[112 * 20];          // 112 rows x 160 bf16 = 35840 B
  unsigned short* lA = reinterpret_cast<unsigned short*>(lAq);

  int tid  = threadIdx.x;
  int lane = tid & 31;
  int wave = tid >> 5;
  int ncol = (blockIdx.x * 8 + wave) * 16 + (lane & 15);
  int hi8  = (lane >> 4) * 8;
  int mlane = lane & 15;

  v8f acc[7];
  #pragma unroll
  for (int t = 0; t < 7; ++t) acc[t] = v8f{0.f,0.f,0.f,0.f,0.f,0.f,0.f,0.f};

  for (int chunk = 0; chunk < 3; ++chunk) {
    __syncthreads();                       // previous chunk fully consumed
    for (int i = tid; i < 112 * 20; i += 256) {
      int m = i / 20, j = i - m * 20;
      const uint4* g = reinterpret_cast<const uint4*>(
          Wb + (size_t)m * 480 + chunk * 160) + j;
#ifdef ASYNC_LDS
      __builtin_amdgcn_global_load_async_to_lds_b128(
          (v4i*)g, (v4i*)(lAq + i), 0, 0);
#else
      lAq[i] = *g;
#endif
    }
#ifdef ASYNC_LDS
#if __has_builtin(__builtin_amdgcn_s_wait_asynccnt)
    __builtin_amdgcn_s_wait_asynccnt(0);
#else
    asm volatile("s_wait_asynccnt 0x0" ::: "memory");
#endif
#endif
    __syncthreads();

    if (chunk < 2 && tid < 112)            // warm L2 for the next weight chunk
      __builtin_prefetch(Wb + (size_t)tid * 480 + (chunk + 1) * 160, 0, 3);

    #pragma unroll
    for (int k5 = 0; k5 < 5; ++k5) {
      int kq = (chunk * 5 + k5) * 4 + (lane >> 4);   // c0/8
      const uint4* bp = reinterpret_cast<const uint4*>(
          Col + ((size_t)kq * Npad + ncol) * 8);
      union { uint4 q[2]; v16bf v; } B;
      B.q[0] = bp[0];                      // K = c0 .. c0+7
      B.q[1] = bp[2 * (size_t)Npad];       // K = c0+16 .. c0+23
      int lc0 = k5 * 32 + hi8;
      #pragma unroll
      for (int t = 0; t < 7; ++t) {
        int m = t * 16 + mlane;
        const uint4* ap = reinterpret_cast<const uint4*>(lA + m * 160 + lc0);
        union { uint4 q[2]; v16bf v; } A;
        A.q[0] = ap[0];
        A.q[1] = ap[2];
        acc[t] = __builtin_amdgcn_wmma_f32_16x16x32_bf16(
            false, A.v, false, B.v, (short)0, acc[t], false, false);
      }
    }
  }

  if (ncol < N) {
    int mrow = (lane >> 4) * 8;
    #pragma unroll
    for (int t = 0; t < 7; ++t)
      #pragma unroll
      for (int r = 0; r < 8; ++r) {
        int m = t * 16 + mrow + r;
        if (m < 100) gates[(size_t)m * N + ncol] = acc[t][r];
      }
  }
}

// ---------------------------------------------------------------------------
// 8) LSTM pointwise: gates [100][N] + bias -> h_next, c_next.
// ---------------------------------------------------------------------------
__global__ void lstm_k(float* __restrict__ outh, float* __restrict__ outc,
                       const float* __restrict__ gates, const float* __restrict__ bias,
                       const float* __restrict__ cin, int szsh, int N) {
  int tid = blockIdx.x * blockDim.x + threadIdx.x;
  int SZ2 = 1 << (2 * szsh);
  if (tid >= 25 * N) return;
  int sp = tid & (SZ2 - 1);
  int bp = tid >> (2 * szsh);
  int p = bp % 25;
  int b = bp / 25;
  int n = b * SZ2 + sp;
  float gi = gates[(size_t)(p)      * N + n] + bias[p];
  float gf = gates[(size_t)(25 + p) * N + n] + bias[25 + p];
  float go = gates[(size_t)(50 + p) * N + n] + bias[50 + p];
  float gg = gates[(size_t)(75 + p) * N + n] + bias[75 + p];
  float ii = 1.f / (1.f + expf(-gi));
  float ff = 1.f / (1.f + expf(-gf));
  float oo = 1.f / (1.f + expf(-go));
  float g  = tanhf(gg);
  float cn = ff * cin[tid] + ii * g;
  outh[tid] = oo * tanhf(cn);
  outc[tid] = cn;
}

// ---------------------------------------------------------------------------
extern "C" void kernel_launch(void* const* d_in, const int* in_sizes, int n_in,
                              void* d_out, int out_size, void* d_ws, size_t ws_size,
                              hipStream_t stream) {
  static const int CHs[6] = {512, 1024, 512, 256, 256, 256};
  static const int SZs[6] = {64, 32, 16, 8, 4, 2};

  // Workspace carve-up (all offsets 16B aligned).
  float* wcomp = (float*)d_ws;                          // 128 f
  float* atab  = wcomp + 128;                           // 1920 f
  float* Ld    = atab + 1920;                           // 73728 f
  unsigned short* feat = (unsigned short*)(Ld + 73728); // 409600 u16
  unsigned short* wb   = feat + 2 * 50 * 4096;          // 53760 u16
  unsigned short* col  = wb + 112 * 480;                // 60*8192*8 = 3932160 u16
  float* gates = (float*)(col + 3932160);               // 100*8192 f

  float* out = (float*)d_out;
  size_t ooff = 0;

  for (int lvl = 0; lvl < 6; ++lvl) {
    int CH = CHs[lvl], SZ = SZs[lvl];
    int szsh = 6 - lvl;
    int lf = lvl + 1;
    int SZ2 = SZ * SZ;
    int N = 2 * SZ2;
    int ng = (N + 127) / 128;
    int Npad = ng * 128;

    const float* xA  = (const float*)d_in[4 * lvl + 0];
    const float* xB  = (const float*)d_in[4 * lvl + 1];
    const float* h   = (const float*)d_in[4 * lvl + 2];
    const float* c   = (const float*)d_in[4 * lvl + 3];
    const float* W   = (const float*)d_in[24 + 2 * lvl];
    const float* bia = (const float*)d_in[25 + 2 * lvl];

    comp_weights_k<<<1, 1, 0, stream>>>(wcomp, lvl + 1);

    int na = 5 * SZ;
    atab_k<<<(na + 63) / 64, 64, 0, stream>>>(atab, wcomp, SZ, lf);

    int nd = 2 * SZ2;
    dot_k<<<(nd + 255) / 256, 256, 0, stream>>>(Ld, xA, xB, CH, SZ, szsh);

    int np = 2 * 50 * SZ2;
    pooled_k<<<(np + 255) / 256, 256, 0, stream>>>(feat, Ld, atab, h, SZ, szsh, np);

    wprep_k<<<(112 * 480 + 255) / 256, 256, 0, stream>>>(wb, W);

    int nq = 60 * Npad;
    pack_k<<<(nq + 255) / 256, 256, 0, stream>>>(col, feat, SZ, szsh, N, Npad);

    gemm_k<<<ng, 256, 0, stream>>>(gates, wb, col, Npad, N);

    int nl = 25 * N;
    lstm_k<<<(nl + 255) / 256, 256, 0, stream>>>(out + ooff, out + ooff + (size_t)25 * N,
                                                 gates, bia, c, szsh, N);
    ooff += (size_t)50 * N;
  }
  (void)in_sizes; (void)n_in; (void)out_size; (void)ws_size;
}
